// Pix2Track_16106127360304
// MI455X (gfx1250) — compile-verified, matrix operand order checked by reading, not run on hardware
//
#include <hip/hip_runtime.h>
#include <hip/hip_bf16.h>
#include <math.h>

// ---------------- Types for WMMA ----------------
typedef __bf16 bf16;
typedef __attribute__((ext_vector_type(8)))  __bf16 bf16x8;
typedef __attribute__((ext_vector_type(16))) __bf16 bf16x16;
typedef __attribute__((ext_vector_type(8)))  float  v8f;

#define HEADS 12
#define DHEAD 64

// ---------------------------------------------------------------------------
// Build memT (L*B, C) = transpose(concat(z,x), (1,0,2)) and
// posIdeT (L*B, C) = pos + identity-broadcast, both in (l*B+b, c) row layout.
// ---------------------------------------------------------------------------
__global__ __launch_bounds__(256)
void build_mem_kernel(const float* __restrict__ z, const float* __restrict__ x,
                      const float* __restrict__ pos_z, const float* __restrict__ pos_x,
                      const float* __restrict__ identity,
                      float* __restrict__ memT, float* __restrict__ posIdeT,
                      int B, int Lz, int Lx, int C)
{
    size_t idx = (size_t)blockIdx.x * 256 + threadIdx.x;
    int L = Lz + Lx;
    size_t total = (size_t)L * B * C;
    if (idx >= total) return;
    int c = (int)(idx % C);
    size_t rb = idx / C;             // l*B + b
    int b = (int)(rb % B);
    int l = (int)(rb / B);
    float m, p, id;
    if (l < Lz) {
        size_t o = ((size_t)b * Lz + l) * C + c;
        m = z[o]; p = pos_z[o]; id = identity[c];
    } else {
        int l2 = l - Lz;
        size_t o = ((size_t)b * Lx + l2) * C + c;
        m = x[o]; p = pos_x[o]; id = identity[C + c];
    }
    memT[idx] = m;
    posIdeT[idx] = p + id;
}

// ---------------------------------------------------------------------------
// Row LayerNorm: Y[row] = LN(X[row]) * g + b  (+ add[row] if add != null)
// One block per row, C = 768.
// ---------------------------------------------------------------------------
__global__ __launch_bounds__(256)
void layernorm_kernel(const float* __restrict__ X,
                      const float* __restrict__ g, const float* __restrict__ bta,
                      const float* __restrict__ add,
                      float* __restrict__ Y, int C)
{
    __shared__ float red[256];
    int row = blockIdx.x;
    int tid = threadIdx.x;
    const float* xr = X + (size_t)row * C;

    float s = 0.f;
    for (int c = tid; c < C; c += 256) s += xr[c];
    red[tid] = s; __syncthreads();
    for (int off = 128; off > 0; off >>= 1) {
        if (tid < off) red[tid] += red[tid + off];
        __syncthreads();
    }
    float mean = red[0] / (float)C;
    __syncthreads();

    float v = 0.f;
    for (int c = tid; c < C; c += 256) { float d = xr[c] - mean; v += d * d; }
    red[tid] = v; __syncthreads();
    for (int off = 128; off > 0; off >>= 1) {
        if (tid < off) red[tid] += red[tid + off];
        __syncthreads();
    }
    float rstd = rsqrtf(red[0] / (float)C + 1e-5f);

    float* yr = Y + (size_t)row * C;
    const float* ar = add ? add + (size_t)row * C : nullptr;
    for (int c = tid; c < C; c += 256) {
        float val = (xr[c] - mean) * rstd * g[c] + bta[c];
        if (ar) val += ar[c];
        yr[c] = val;
    }
}

// ---------------------------------------------------------------------------
// GEMM: Cout[M,N] = act(A[M,K] @ W[N,K]^T + bias) (+ Res)
// Block = 256 threads (8 waves), 128x64 output tile, LDS double-buffered.
// Wave w: rows (w>>1)*32 .. +31, cols (w&1)*32 .. +31  -> 4 WMMA accumulators.
// Per K-step(32): prefetch next slab (global->regs), 4 x v_wmma on current LDS
// buffer, convert+store prefetch to other buffer, ONE barrier.
// Fragments follow the CDNA5 16-bit A/B layout
// (lanes 0-15: K {0..7,16..23}; lanes 16-31: K {8..15,24..31}).
// Requirements: M % 128 == 0, N % 64 == 0, K % 32 == 0 (all call sites comply).
// ---------------------------------------------------------------------------
__global__ __launch_bounds__(256)
void gemm_wmma_kernel(const float* __restrict__ A, const float* __restrict__ W,
                      const float* __restrict__ bias, const float* __restrict__ Res,
                      float* __restrict__ Cout, int M, int N, int K, int gelu)
{
    __shared__ bf16 As[2][128 * 32];
    __shared__ bf16 Bs[2][64 * 32];

    int tid  = threadIdx.x;
    int wave = tid >> 5;
    int lane = tid & 31;
    int lr   = lane & 15;
    int hi   = lane >> 4;                 // 0 or 1

    int mBlk = blockIdx.y * 128;
    int nBlk = blockIdx.x * 64;
    int m0   = (wave >> 1) * 32;          // 0,32,64,96
    int n0   = (wave & 1) * 32;           // 0,32

    v8f acc00 = {}, acc01 = {}, acc10 = {}, acc11 = {};

    // cooperative-load mapping
    int arow = tid >> 1;                  // 0..127
    int acol = (tid & 1) * 16;            // 0 or 16
    int brow = tid >> 2;                  // 0..63
    int bcol = (tid & 3) * 8;             // 0,8,16,24

    // stage first K-slab into buffer 0
    {
        const float* sa = A + (size_t)(mBlk + arow) * K + acol;
        bf16* da = &As[0][arow * 32 + acol];
        #pragma unroll
        for (int j = 0; j < 16; ++j) da[j] = (bf16)sa[j];
        const float* sb = W + (size_t)(nBlk + brow) * K + bcol;
        bf16* db = &Bs[0][brow * 32 + bcol];
        #pragma unroll
        for (int j = 0; j < 8; ++j) db[j] = (bf16)sb[j];
    }
    __syncthreads();

    int nsteps = K >> 5;
    for (int s = 0; s < nsteps; ++s) {
        int cur = s & 1, nxt = cur ^ 1;
        bool pre = (s + 1 < nsteps);
        float pa[16], pb[8];
        if (pre) {                        // prefetch next slab into registers
            int kk = (s + 1) << 5;
            const float* sa = A + (size_t)(mBlk + arow) * K + kk + acol;
            #pragma unroll
            for (int j = 0; j < 16; ++j) pa[j] = sa[j];
            const float* sb = W + (size_t)(nBlk + brow) * K + kk + bcol;
            #pragma unroll
            for (int j = 0; j < 8; ++j) pb[j] = sb[j];
        }

        const bf16* ap0 = &As[cur][(m0 + lr) * 32 + hi * 8];
        const bf16* ap1 = ap0 + 16 * 32;
        const bf16* bp0 = &Bs[cur][(n0 + lr) * 32 + hi * 8];
        const bf16* bp1 = bp0 + 16 * 32;

        bf16x8 a0lo = *(const bf16x8*)ap0;
        bf16x8 a0hi = *(const bf16x8*)(ap0 + 16);
        bf16x8 a1lo = *(const bf16x8*)ap1;
        bf16x8 a1hi = *(const bf16x8*)(ap1 + 16);
        bf16x8 b0lo = *(const bf16x8*)bp0;
        bf16x8 b0hi = *(const bf16x8*)(bp0 + 16);
        bf16x8 b1lo = *(const bf16x8*)bp1;
        bf16x8 b1hi = *(const bf16x8*)(bp1 + 16);

        bf16x16 af0 = __builtin_shufflevector(a0lo, a0hi, 0,1,2,3,4,5,6,7,8,9,10,11,12,13,14,15);
        bf16x16 af1 = __builtin_shufflevector(a1lo, a1hi, 0,1,2,3,4,5,6,7,8,9,10,11,12,13,14,15);
        bf16x16 bf0 = __builtin_shufflevector(b0lo, b0hi, 0,1,2,3,4,5,6,7,8,9,10,11,12,13,14,15);
        bf16x16 bf1 = __builtin_shufflevector(b1lo, b1hi, 0,1,2,3,4,5,6,7,8,9,10,11,12,13,14,15);

        acc00 = __builtin_amdgcn_wmma_f32_16x16x32_bf16(false, af0, false, bf0, (short)0, acc00, false, false);
        acc01 = __builtin_amdgcn_wmma_f32_16x16x32_bf16(false, af0, false, bf1, (short)0, acc01, false, false);
        acc10 = __builtin_amdgcn_wmma_f32_16x16x32_bf16(false, af1, false, bf0, (short)0, acc10, false, false);
        acc11 = __builtin_amdgcn_wmma_f32_16x16x32_bf16(false, af1, false, bf1, (short)0, acc11, false, false);

        if (pre) {                        // convert + store prefetch to other buffer
            bf16* da = &As[nxt][arow * 32 + acol];
            #pragma unroll
            for (int j = 0; j < 16; ++j) da[j] = (bf16)pa[j];
            bf16* db = &Bs[nxt][brow * 32 + bcol];
            #pragma unroll
            for (int j = 0; j < 8; ++j) db[j] = (bf16)pb[j];
        }
        __syncthreads();
    }

    // Epilogue: D layout -> lane lr = N col; VGPR r = row (r + hi*8) in tile.
    int col0 = nBlk + n0 + lr;
    int col1 = col0 + 16;
    float bv0 = bias ? bias[col0] : 0.f;
    float bv1 = bias ? bias[col1] : 0.f;
    #pragma unroll
    for (int r = 0; r < 8; ++r) {
        int row0 = mBlk + m0 + r + hi * 8;
        int row1 = row0 + 16;
        float v00 = acc00[r] + bv0;
        float v01 = acc01[r] + bv1;
        float v10 = acc10[r] + bv0;
        float v11 = acc11[r] + bv1;
        if (gelu) {
            v00 = 0.5f * v00 * (1.f + erff(v00 * 0.70710678118f));
            v01 = 0.5f * v01 * (1.f + erff(v01 * 0.70710678118f));
            v10 = 0.5f * v10 * (1.f + erff(v10 * 0.70710678118f));
            v11 = 0.5f * v11 * (1.f + erff(v11 * 0.70710678118f));
        }
        if (Res) {
            v00 += Res[(size_t)row0 * N + col0];
            v01 += Res[(size_t)row0 * N + col1];
            v10 += Res[(size_t)row1 * N + col0];
            v11 += Res[(size_t)row1 * N + col1];
        }
        Cout[(size_t)row0 * N + col0] = v00;
        Cout[(size_t)row0 * N + col1] = v01;
        Cout[(size_t)row1 * N + col0] = v10;
        Cout[(size_t)row1 * N + col1] = v11;
    }
}

// ---------------------------------------------------------------------------
// Attention core: one block per (b, h). Q rows are q*B+b (seq-first layout),
// K/V rows are k*B+b. Computes O = softmax(Q K^T * 1/8 [+ causal]) V.
// Lq = 32, Lk <= 320. ~2.5% of total FLOPs -> VALU is fine here.
// ---------------------------------------------------------------------------
__global__ __launch_bounds__(256)
void attention_kernel(const float* __restrict__ Qb, int ldQ,
                      const float* __restrict__ Kb, int ldK,
                      const float* __restrict__ Vb, int ldV,
                      float* __restrict__ Ob, int ldO,
                      int Bsz, int Lq, int Lk, int causal)
{
    __shared__ float S[32 * 320];
    __shared__ float Qs[32 * DHEAD];

    int bh = blockIdx.x;
    int b = bh / HEADS, h = bh % HEADS;
    int tid = threadIdx.x;
    const float scale = 0.125f;   // 1/sqrt(64)

    for (int idx = tid; idx < Lq * DHEAD; idx += 256) {
        int q = idx / DHEAD, d = idx % DHEAD;
        Qs[q * DHEAD + d] = Qb[(size_t)(q * Bsz + b) * ldQ + h * DHEAD + d];
    }
    __syncthreads();

    for (int idx = tid; idx < Lq * Lk; idx += 256) {
        int q = idx / Lk, k = idx % Lk;
        float s;
        if (causal && k > q) {
            s = -__builtin_inff();
        } else {
            const float* kp = Kb + (size_t)(k * Bsz + b) * ldK + h * DHEAD;
            float acc = 0.f;
            #pragma unroll 8
            for (int d = 0; d < DHEAD; ++d) acc += Qs[q * DHEAD + d] * kp[d];
            s = acc * scale;
        }
        S[q * Lk + k] = s;
    }
    __syncthreads();

    if (tid < Lq) {
        int q = tid;
        float m = -__builtin_inff();
        for (int k = 0; k < Lk; ++k) m = fmaxf(m, S[q * Lk + k]);
        float sum = 0.f;
        for (int k = 0; k < Lk; ++k) {
            float e = __expf(S[q * Lk + k] - m);
            S[q * Lk + k] = e;
            sum += e;
        }
        float inv = 1.f / sum;
        for (int k = 0; k < Lk; ++k) S[q * Lk + k] *= inv;
    }
    __syncthreads();

    for (int idx = tid; idx < Lq * DHEAD; idx += 256) {
        int q = idx / DHEAD, d = idx % DHEAD;
        float acc = 0.f;
        for (int k = 0; k < Lk; ++k)
            acc += S[q * Lk + k] * Vb[(size_t)(k * Bsz + b) * ldV + h * DHEAD + d];
        Ob[(size_t)(q * Bsz + b) * ldO + h * DHEAD + d] = acc;
    }
}

// ---------------------------------------------------------------------------
// Host orchestration
// ---------------------------------------------------------------------------
extern "C" void kernel_launch(void* const* d_in, const int* in_sizes, int n_in,
                              void* d_out, int out_size, void* d_ws, size_t ws_size,
                              hipStream_t stream)
{
    (void)in_sizes; (void)n_in; (void)out_size; (void)ws_size;

    const int C = 768, Hid = 3072, B = 64, Nq = 32, Lz = 64, Lx = 256, NL = 6;
    const int L = Lz + Lx;                 // 320
    const int Mq = Nq * B;                 // 2048
    const int Mm = L * B;                  // 20480

    const float* tgt       = (const float*)d_in[0];
    const float* z         = (const float*)d_in[1];
    const float* x         = (const float*)d_in[2];
    const float* pos_z     = (const float*)d_in[3];
    const float* pos_x     = (const float*)d_in[4];
    const float* identity  = (const float*)d_in[5];
    const float* query_pos = (const float*)d_in[6];
    const float* n1_g  = (const float*)d_in[7];
    const float* n1_b  = (const float*)d_in[8];
    const float* sa_w  = (const float*)d_in[9];
    const float* sa_b  = (const float*)d_in[10];
    const float* sa_ow = (const float*)d_in[11];
    const float* sa_ob = (const float*)d_in[12];
    const float* n2q_g = (const float*)d_in[13];
    const float* n2q_b = (const float*)d_in[14];
    const float* n2m_g = (const float*)d_in[15];
    const float* n2m_b = (const float*)d_in[16];
    const float* ca_w  = (const float*)d_in[17];
    const float* ca_b  = (const float*)d_in[18];
    const float* ca_ow = (const float*)d_in[19];
    const float* ca_ob = (const float*)d_in[20];
    const float* n3_g  = (const float*)d_in[21];
    const float* n3_b  = (const float*)d_in[22];
    const float* fc1_w = (const float*)d_in[23];
    const float* fc1_b = (const float*)d_in[24];
    const float* fc2_w = (const float*)d_in[25];
    const float* fc2_b = (const float*)d_in[26];
    const float* fn_g  = (const float*)d_in[27];
    const float* fn_b  = (const float*)d_in[28];

    // Workspace carve-up (floats). Buffers time-shared where lifetimes allow.
    float* ws = (float*)d_ws;
    float* q_out   = ws;                 ws += (size_t)Mq * C;       // residual stream
    float* buf_a   = ws;                 ws += (size_t)Mq * C;       // normed q / MLP in
    float* attn    = ws;                 ws += (size_t)Mq * C;       // attention output
    float* qproj   = ws;                 ws += (size_t)Mq * C;       // sa_v  OR ca_q
    float* big     = ws;                 ws += (size_t)Mq * Hid;     // sa_qk OR mlp hidden
    float* memT    = ws;                 ws += (size_t)Mm * C;
    float* posIdeT = ws;                 ws += (size_t)Mm * C;
    float* k2v     = ws;                 ws += (size_t)Mm * C;       // k2 then ca_v
    float* ca_k    = ws;                 ws += (size_t)Mm * C;

    // q_out = tgt
    hipMemcpyAsync(q_out, tgt, (size_t)Mq * C * sizeof(float),
                   hipMemcpyDeviceToDevice, stream);

    // memT / posIdeT (loop-invariant)
    {
        size_t total = (size_t)Mm * C;
        int blocks = (int)((total + 255) / 256);
        build_mem_kernel<<<blocks, 256, 0, stream>>>(z, x, pos_z, pos_x, identity,
                                                     memT, posIdeT, B, Lz, Lx, C);
    }

    dim3 blk(256);
    dim3 gq768(C / 64, Mq / 128);          // (12, 16)
    dim3 gq1536(2 * C / 64, Mq / 128);     // (24, 16)
    dim3 gq3072(Hid / 64, Mq / 128);       // (48, 16)
    dim3 gm768(C / 64, Mm / 128);          // (12, 160)

    for (int i = 0; i < NL; ++i) {
        const float* sa_w_i  = sa_w  + (size_t)i * 3 * C * C;
        const float* sa_b_i  = sa_b  + (size_t)i * 3 * C;
        const float* sa_ow_i = sa_ow + (size_t)i * C * C;
        const float* sa_ob_i = sa_ob + (size_t)i * C;
        const float* ca_w_i  = ca_w  + (size_t)i * 3 * C * C;
        const float* ca_b_i  = ca_b  + (size_t)i * 3 * C;
        const float* ca_ow_i = ca_ow + (size_t)i * C * C;
        const float* ca_ob_i = ca_ob + (size_t)i * C;
        const float* fc1_w_i = fc1_w + (size_t)i * Hid * C;
        const float* fc1_b_i = fc1_b + (size_t)i * Hid;
        const float* fc2_w_i = fc2_w + (size_t)i * C * Hid;
        const float* fc2_b_i = fc2_b + (size_t)i * C;

        // ---- Self-attention (pre-norm; V taken from un-normed residual) ----
        layernorm_kernel<<<Mq, blk, 0, stream>>>(q_out, n1_g + i * C, n1_b + i * C,
                                                 query_pos, buf_a, C);
        // QK projection (N = 2C)
        gemm_wmma_kernel<<<gq1536, blk, 0, stream>>>(buf_a, sa_w_i, sa_b_i, nullptr,
                                                     big, Mq, 2 * C, C, 0);
        // V projection from q_out
        gemm_wmma_kernel<<<gq768, blk, 0, stream>>>(q_out, sa_w_i + (size_t)2 * C * C,
                                                    sa_b_i + 2 * C, nullptr,
                                                    qproj, Mq, C, C, 0);
        attention_kernel<<<B * HEADS, blk, 0, stream>>>(big, 2 * C, big + C, 2 * C,
                                                        qproj, C, attn, C,
                                                        B, Nq, Nq, 1);
        // out-proj + residual (in place on q_out)
        gemm_wmma_kernel<<<gq768, blk, 0, stream>>>(attn, sa_ow_i, sa_ob_i, q_out,
                                                    q_out, Mq, C, C, 0);

        // ---- Cross-attention ----
        layernorm_kernel<<<Mq, blk, 0, stream>>>(q_out, n2q_g + i * C, n2q_b + i * C,
                                                 query_pos, buf_a, C);
        layernorm_kernel<<<Mm, blk, 0, stream>>>(memT, n2m_g + i * C, n2m_b + i * C,
                                                 posIdeT, k2v, C);
        gemm_wmma_kernel<<<gq768, blk, 0, stream>>>(buf_a, ca_w_i, ca_b_i, nullptr,
                                                    qproj, Mq, C, C, 0);
        gemm_wmma_kernel<<<gm768, blk, 0, stream>>>(k2v, ca_w_i + (size_t)C * C,
                                                    ca_b_i + C, nullptr,
                                                    ca_k, Mm, C, C, 0);
        gemm_wmma_kernel<<<gm768, blk, 0, stream>>>(memT, ca_w_i + (size_t)2 * C * C,
                                                    ca_b_i + 2 * C, nullptr,
                                                    k2v, Mm, C, C, 0);   // k2v now = V
        attention_kernel<<<B * HEADS, blk, 0, stream>>>(qproj, C, ca_k, C,
                                                        k2v, C, attn, C,
                                                        B, Nq, L, 0);
        gemm_wmma_kernel<<<gq768, blk, 0, stream>>>(attn, ca_ow_i, ca_ob_i, q_out,
                                                    q_out, Mq, C, C, 0);

        // ---- MLP (exact GELU) ----
        layernorm_kernel<<<Mq, blk, 0, stream>>>(q_out, n3_g + i * C, n3_b + i * C,
                                                 nullptr, buf_a, C);
        gemm_wmma_kernel<<<gq3072, blk, 0, stream>>>(buf_a, fc1_w_i, fc1_b_i, nullptr,
                                                     big, Mq, Hid, C, 1);
        gemm_wmma_kernel<<<gq768, blk, 0, stream>>>(big, fc2_w_i, fc2_b_i, q_out,
                                                    q_out, Mq, C, Hid, 0);
    }

    // Final LayerNorm -> d_out (Nq, B, C) == rows q*B+b
    layernorm_kernel<<<Mq, blk, 0, stream>>>(q_out, fn_g, fn_b, nullptr,
                                             (float*)d_out, C);
}
